// MultiHeadAttention_54863912239403
// MI455X (gfx1250) — compile-verified, matrix operand order checked by reading, not run on hardware
//
#include <hip/hip_runtime.h>

#define DEV __device__ __forceinline__

typedef __attribute__((ext_vector_type(16))) _Float16 v16h;
typedef __attribute__((ext_vector_type(8)))  float    v8f;
typedef __attribute__((ext_vector_type(4)))  int      v4i;

union FragU {
  uint4 u[2];
  v16h  h;
};

static constexpr int B_  = 2;
static constexpr int S_  = 2048;
static constexpr int D_  = 1024;
static constexpr int H_  = 16;
static constexpr int DH_ = 64;
static constexpr int M_  = B_ * S_;   // 4096 rows for the projection GEMMs
static constexpr int LDK = 40;        // padded K stride (f16 elems) in LDS: 80B rows, 16B aligned

#define HAS_ASYNC_LDS  __has_builtin(__builtin_amdgcn_global_load_async_to_lds_b128)

DEV v8f wmma_f16(const FragU& a, const FragU& b, v8f c) {
  return __builtin_amdgcn_wmma_f32_16x16x32_f16(false, a.h, false, b.h,
                                                (short)0, c, false, false);
}

// 16-byte global -> LDS copy. Prefer the CDNA5 async pipe (ASYNCcnt, no VGPR
// round trip); fall back to a synchronous copy if the builtin is absent.
DEV void copy16_g2l(const _Float16* g, _Float16* l) {
#if HAS_ASYNC_LDS
  __builtin_amdgcn_global_load_async_to_lds_b128(
      (__attribute__((address_space(1))) v4i*)g,
      (__attribute__((address_space(3))) v4i*)l, 0, 0);
#else
  *reinterpret_cast<uint4*>(l) = *reinterpret_cast<const uint4*>(g);
#endif
}

DEV void wait_async_copies() {
#if HAS_ASYNC_LDS
# if __has_builtin(__builtin_amdgcn_s_wait_asynccnt)
  __builtin_amdgcn_s_wait_asynccnt(0);
# else
  asm volatile("s_wait_asynccnt 0" ::: "memory");
# endif
#endif
}

// ---------------- one-time precision / layout conversion ----------------

// f32 -> f16 elementwise, 4 elems per thread.
__global__ __launch_bounds__(256)
void cvt_f16_kernel(const float* __restrict__ src, _Float16* __restrict__ dst)
{
  size_t i = ((size_t)blockIdx.x * 256 + threadIdx.x) * 4;
  float4 v = *reinterpret_cast<const float4*>(src + i);
  union { _Float16 h[4]; uint2 u; } cv;
  cv.h[0] = (_Float16)v.x; cv.h[1] = (_Float16)v.y;
  cv.h[2] = (_Float16)v.z; cv.h[3] = (_Float16)v.w;
  *reinterpret_cast<uint2*>(dst + i) = cv.u;
}

// W f32 [K=1024][N=1024] -> Wt f16 [N][K] (transpose + convert), 64x64 tiles.
__global__ __launch_bounds__(256)
void cvt_wt_kernel(const float* __restrict__ W, _Float16* __restrict__ Wt)
{
  __shared__ _Float16 t[64 * 72];
  const int tid = threadIdx.x;
  const int bk  = blockIdx.x & 15;
  const int bn  = blockIdx.x >> 4;
  #pragma unroll
  for (int it = 0; it < 4; ++it) {
    int e = it * 1024 + tid * 4;
    int r = e >> 6, c = e & 63;
    float4 v = *reinterpret_cast<const float4*>(W + (size_t)(bk * 64 + r) * 1024 + bn * 64 + c);
    t[(c + 0) * 72 + r] = (_Float16)v.x;
    t[(c + 1) * 72 + r] = (_Float16)v.y;
    t[(c + 2) * 72 + r] = (_Float16)v.z;
    t[(c + 3) * 72 + r] = (_Float16)v.w;
  }
  __syncthreads();
  #pragma unroll
  for (int it = 0; it < 4; ++it) {
    int e = it * 1024 + tid * 4;
    int n = e >> 6, k = e & 63;
    *reinterpret_cast<uint2*>(Wt + (size_t)(bn * 64 + n) * 1024 + bk * 64 + k) =
        *reinterpret_cast<const uint2*>(&t[n * 72 + k]);
  }
}

// ---------------- WMMA GEMM: C[M,N] = A[M,K] * Bt[N,K]^T + bias ----------------
// A f16 [M,1024], Bt f16 [N][K] (pre-transposed). 128x128 block tile, K-step 32,
// 8 waves, wave tile 32x64. Double-buffered async global->LDS staging.
// OUT_MODE: 0 = f16 [M,N], 1 = f16 V-transposed [B,H,DH,S], 2 = f32 [M,N]
template <int OUT_MODE>
__global__ __launch_bounds__(256)
void gemm_kernel(const _Float16* __restrict__ A, const _Float16* __restrict__ Bt,
                 const float* __restrict__ bias, void* __restrict__ Cout)
{
  __shared__ _Float16 As[2][128 * LDK];   // [m][k], double buffered
  __shared__ _Float16 Bs[2][128 * LDK];   // [n][k], double buffered

  const int tid  = threadIdx.x;
  const int lane = tid & 31;
  const int wave = tid >> 5;
  const int bm   = blockIdx.x >> 3;
  const int bn   = blockIdx.x & 7;
  const int wm   = wave & 3;
  const int wn   = wave >> 2;
  const int lr   = lane & 15;
  const int kof  = (lane >> 4) * 8;

  const _Float16* Ab = A  + (size_t)(bm * 128) * 1024;
  const _Float16* Bb = Bt + (size_t)(bn * 128) * 1024;

  // per-thread chunk mapping: 512 16B chunks per 128x32 tile, 2 per thread
  int c0 = tid, c1 = tid + 256;
  int ar0 = c0 >> 2, ak0 = (c0 & 3) * 8;
  int ar1 = c1 >> 2, ak1 = (c1 & 3) * 8;

  v8f acc[2][4] = {};

  // prologue: stage tile 0 into buffer 0
  copy16_g2l(Ab + (size_t)ar0 * 1024 + ak0, &As[0][ar0 * LDK + ak0]);
  copy16_g2l(Ab + (size_t)ar1 * 1024 + ak1, &As[0][ar1 * LDK + ak1]);
  copy16_g2l(Bb + (size_t)ar0 * 1024 + ak0, &Bs[0][ar0 * LDK + ak0]);
  copy16_g2l(Bb + (size_t)ar1 * 1024 + ak1, &Bs[0][ar1 * LDK + ak1]);

  for (int kt = 0; kt < 1024 / 32; ++kt) {
    wait_async_copies();
    __syncthreads();

    const int cur = kt & 1;
    if (kt + 1 < 1024 / 32) {   // stage next tile while this one computes
      const int nxt = cur ^ 1;
      const size_t go = (size_t)(kt + 1) * 32;
      copy16_g2l(Ab + (size_t)ar0 * 1024 + go + ak0, &As[nxt][ar0 * LDK + ak0]);
      copy16_g2l(Ab + (size_t)ar1 * 1024 + go + ak1, &As[nxt][ar1 * LDK + ak1]);
      copy16_g2l(Bb + (size_t)ar0 * 1024 + go + ak0, &Bs[nxt][ar0 * LDK + ak0]);
      copy16_g2l(Bb + (size_t)ar1 * 1024 + go + ak1, &Bs[nxt][ar1 * LDK + ak1]);
    }

    FragU af[2], bf[4];
    #pragma unroll
    for (int mt = 0; mt < 2; ++mt) {
      int r = wm * 32 + mt * 16 + lr;
      af[mt].u[0] = *reinterpret_cast<const uint4*>(&As[cur][r * LDK + kof]);
      af[mt].u[1] = *reinterpret_cast<const uint4*>(&As[cur][r * LDK + 16 + kof]);
    }
    #pragma unroll
    for (int nt = 0; nt < 4; ++nt) {
      int c = wn * 64 + nt * 16 + lr;
      bf[nt].u[0] = *reinterpret_cast<const uint4*>(&Bs[cur][c * LDK + kof]);
      bf[nt].u[1] = *reinterpret_cast<const uint4*>(&Bs[cur][c * LDK + 16 + kof]);
    }
    #pragma unroll
    for (int mt = 0; mt < 2; ++mt)
      #pragma unroll
      for (int nt = 0; nt < 4; ++nt)
        acc[mt][nt] = wmma_f16(af[mt], bf[nt], acc[mt][nt]);
  }

  // ---- epilogue: bias + store ----
  #pragma unroll
  for (int mt = 0; mt < 2; ++mt) {
    #pragma unroll
    for (int nt = 0; nt < 4; ++nt) {
      int rg = bm * 128 + wm * 32 + mt * 16 + (lane >> 4) * 8;
      int cg = bn * 128 + wn * 64 + nt * 16 + lr;
      float bv = bias[cg];
      #pragma unroll
      for (int i = 0; i < 8; ++i) {
        float val = acc[mt][nt][i] + bv;
        int r = rg + i;
        if (OUT_MODE == 2) {
          reinterpret_cast<float*>(Cout)[(size_t)r * 1024 + cg] = val;
        } else if (OUT_MODE == 0) {
          reinterpret_cast<_Float16*>(Cout)[(size_t)r * 1024 + cg] = (_Float16)val;
        } else {  // V stored transposed: [B, H, DH, S]
          int b = r >> 11, s = r & 2047;
          int h = cg >> 6, dh = cg & 63;
          reinterpret_cast<_Float16*>(Cout)[(((size_t)(b * H_ + h)) * DH_ + dh) * S_ + s] =
              (_Float16)val;
        }
      }
    }
  }
}

// ---------------- flash attention ----------------
// One wave per 16-query tile; 32 keys per step; causal mask applied analytically.
// Q,K: f16 [B,S,D] (head cols h*64+dh).  Vt: f16 [B,H,DH,S].  O: f16 [B,S,D].
__global__ __launch_bounds__(256)
void attn_kernel(const _Float16* __restrict__ Q, const _Float16* __restrict__ K,
                 const _Float16* __restrict__ Vt, _Float16* __restrict__ O)
{
  __shared__ _Float16 Pl[8][16 * 32];  // per-wave P repack buffer (C-layout -> A-layout)

  const int tid  = threadIdx.x;
  const int lane = tid & 31;
  const int wave = tid >> 5;
  int blk = blockIdx.x;                       // grid = B*H*(S/128)
  const int qg = blk & 15;  blk >>= 4;
  const int h  = blk & 15;  blk >>= 4;
  const int b  = blk;
  const int qbase = qg * 128 + wave * 16;

  const size_t qkBase = (size_t)b * S_ * D_ + (size_t)h * DH_;
  const size_t vBase  = (size_t)(b * H_ + h) * DH_ * S_;
  const int lr  = lane & 15;
  const int hh  = lane >> 4;
  const int kof = hh * 8;

  FragU qf[2];
  #pragma unroll
  for (int f = 0; f < 2; ++f) {
    const _Float16* qp = Q + qkBase + (size_t)(qbase + lr) * D_ + f * 32 + kof;
    qf[f].u[0] = *reinterpret_cast<const uint4*>(qp);
    qf[f].u[1] = *reinterpret_cast<const uint4*>(qp + 16);
  }

  v8f o[4] = {};
  float mrow[8], lrow[8];
  #pragma unroll
  for (int i = 0; i < 8; ++i) { mrow[i] = -1e30f; lrow[i] = 0.0f; }

  const int kbmax = (qbase + 15) >> 5;
  for (int kb = 0; kb <= kbmax; ++kb) {
    v8f s0 = {}, s1 = {};
    #pragma unroll
    for (int f = 0; f < 2; ++f) {
      FragU kf0, kf1;
      const _Float16* kp0 = K + qkBase + (size_t)(kb * 32 + lr) * D_ + f * 32 + kof;
      const _Float16* kp1 = kp0 + (size_t)16 * D_;
      kf0.u[0] = *reinterpret_cast<const uint4*>(kp0);
      kf0.u[1] = *reinterpret_cast<const uint4*>(kp0 + 16);
      kf1.u[0] = *reinterpret_cast<const uint4*>(kp1);
      kf1.u[1] = *reinterpret_cast<const uint4*>(kp1 + 16);
      s0 = wmma_f16(qf[f], kf0, s0);
      s1 = wmma_f16(qf[f], kf1, s1);
    }

    float p0[8], p1[8], fac[8];
    #pragma unroll
    for (int i = 0; i < 8; ++i) {
      int row  = i + (hh << 3);
      int qidx = qbase + row;
      int k0   = kb * 32 + lr, k1 = k0 + 16;
      float a = s0[i] * 0.125f;  if (k0 > qidx) a = -1e30f;   // 1/sqrt(DH)
      float c = s1[i] * 0.125f;  if (k1 > qidx) c = -1e30f;
      float mx = fmaxf(a, c);
      mx = fmaxf(mx, __shfl_xor(mx, 1));
      mx = fmaxf(mx, __shfl_xor(mx, 2));
      mx = fmaxf(mx, __shfl_xor(mx, 4));
      mx = fmaxf(mx, __shfl_xor(mx, 8));
      float mnew = fmaxf(mrow[i], mx);
      fac[i] = __expf(mrow[i] - mnew);
      p0[i]  = __expf(a - mnew);
      p1[i]  = __expf(c - mnew);
      float rs = p0[i] + p1[i];
      rs += __shfl_xor(rs, 1);
      rs += __shfl_xor(rs, 2);
      rs += __shfl_xor(rs, 4);
      rs += __shfl_xor(rs, 8);
      lrow[i] = lrow[i] * fac[i] + rs;
      mrow[i] = mnew;
    }
    #pragma unroll
    for (int dt = 0; dt < 4; ++dt)
      #pragma unroll
      for (int i = 0; i < 8; ++i) o[dt][i] *= fac[i];

    _Float16* pl = &Pl[wave][0];
    #pragma unroll
    for (int i = 0; i < 8; ++i) {
      int row = i + (hh << 3);
      pl[row * 32 + lr]      = (_Float16)p0[i];
      pl[row * 32 + 16 + lr] = (_Float16)p1[i];
    }
    FragU pf;
    pf.u[0] = *reinterpret_cast<const uint4*>(&pl[lr * 32 + kof]);
    pf.u[1] = *reinterpret_cast<const uint4*>(&pl[lr * 32 + 16 + kof]);

    #pragma unroll
    for (int dt = 0; dt < 4; ++dt) {
      FragU vf;
      const _Float16* vp = Vt + vBase + (size_t)(dt * 16 + lr) * S_ + kb * 32 + kof;
      vf.u[0] = *reinterpret_cast<const uint4*>(vp);
      vf.u[1] = *reinterpret_cast<const uint4*>(vp + 16);
      o[dt] = wmma_f16(pf, vf, o[dt]);
    }
  }

  float inv[8];
  #pragma unroll
  for (int i = 0; i < 8; ++i) inv[i] = 1.0f / lrow[i];
  #pragma unroll
  for (int dt = 0; dt < 4; ++dt)
    #pragma unroll
    for (int i = 0; i < 8; ++i) {
      int row = i + (hh << 3);
      O[qkBase + (size_t)(qbase + row) * D_ + dt * 16 + lr] = (_Float16)(o[dt][i] * inv[i]);
    }
}

extern "C" void kernel_launch(void* const* d_in, const int* in_sizes, int n_in,
                              void* d_out, int out_size, void* d_ws, size_t ws_size,
                              hipStream_t stream)
{
  (void)in_sizes; (void)n_in; (void)out_size; (void)ws_size;
  const float* x  = (const float*)d_in[0];
  // d_in[1] is the tril mask; causality is applied analytically in attn_kernel.
  const float* Wq = (const float*)d_in[2];
  const float* bq = (const float*)d_in[3];
  const float* Wk = (const float*)d_in[4];
  const float* bk = (const float*)d_in[5];
  const float* Wv = (const float*)d_in[6];
  const float* bv = (const float*)d_in[7];
  const float* Wo = (const float*)d_in[8];
  const float* bo = (const float*)d_in[9];

  // workspace layout (f16): xh 8MB | 4x Wt 2MB | Q,K,Vt,attn 8MB each = 48MB
  _Float16* xh  = (_Float16*)d_ws;
  _Float16* Wqt = xh  + (size_t)M_ * D_;
  _Float16* Wkt = Wqt + (size_t)D_ * D_;
  _Float16* Wvt = Wkt + (size_t)D_ * D_;
  _Float16* Wot = Wvt + (size_t)D_ * D_;
  _Float16* Qws = Wot + (size_t)D_ * D_;
  _Float16* Kws = Qws + (size_t)M_ * D_;
  _Float16* Vws = Kws + (size_t)M_ * D_;
  _Float16* Aws = Vws + (size_t)M_ * D_;

  dim3 blk(256);
  cvt_f16_kernel<<<dim3((M_ * D_) / 1024), blk, 0, stream>>>(x, xh);
  cvt_wt_kernel <<<dim3(256), blk, 0, stream>>>(Wq, Wqt);
  cvt_wt_kernel <<<dim3(256), blk, 0, stream>>>(Wk, Wkt);
  cvt_wt_kernel <<<dim3(256), blk, 0, stream>>>(Wv, Wvt);
  cvt_wt_kernel <<<dim3(256), blk, 0, stream>>>(Wo, Wot);

  dim3 gg(256);
  gemm_kernel<0><<<gg, blk, 0, stream>>>(xh, Wqt, bq, (void*)Qws);
  gemm_kernel<0><<<gg, blk, 0, stream>>>(xh, Wkt, bk, (void*)Kws);
  gemm_kernel<1><<<gg, blk, 0, stream>>>(xh, Wvt, bv, (void*)Vws);
  attn_kernel<<<dim3(B_ * H_ * (S_ / 128)), blk, 0, stream>>>(Qws, Kws, Vws, Aws);
  gemm_kernel<2><<<gg, blk, 0, stream>>>(Aws, Wot, bo, d_out);
}
